// AudioAttention_65936337928620
// MI455X (gfx1250) — matrix operand order repaired and checked by reading, not compiled, not run on hardware
//
#include <hip/hip_runtime.h>
#include <hip/hip_bf16.h>

// ---------------------------------------------------------------------------
// Gemma audio chunked attention on MI455X (gfx1250, wave32, WMMA + TDM).
// Big GEMMs: bf16 WMMA (v_wmma_f32_16x16x32_bf16, f32 accum), tiles staged
// into LDS by the Tensor Data Mover (tensor_load_to_lds), double-buffered.
// ---------------------------------------------------------------------------

typedef __attribute__((ext_vector_type(8)))  float    v8f;
typedef __attribute__((ext_vector_type(16))) __bf16   v16bf;
typedef __attribute__((ext_vector_type(4)))  unsigned v4u;
typedef __attribute__((ext_vector_type(8)))  int      v8i;
typedef __attribute__((ext_vector_type(4)))  int      v4i;

#define S_LEN   6144
#define NHEAD   8
#define DHEAD   128
#define HIDDIM  1024
#define CHUNK   12
#define CTX     24
#define NB      512          // S_LEN / CHUNK (exact, no padding)
#define BATCH   4
#define MTOT    (BATCH * S_LEN)   // 24576 rows

static constexpr float QSC = 0.08838834764831845f / 0.6931471805599453f; // D^-.5/ln2
static constexpr float KSC = 1.3132616875182228f  / 0.6931471805599453f; // ln(1+e)/ln2

__device__ __forceinline__ unsigned short f2b(float f) {
    unsigned u = __float_as_uint(f);
    unsigned r = (u + 0x7FFFu + ((u >> 16) & 1u)) >> 16;
    return (unsigned short)r;
}

__device__ __forceinline__ v16bf mkfrag(uint4 a, uint4 b) {
    union { uint4 u[2]; v16bf v; } x;
    x.u[0] = a; x.u[1] = b;
    return x.v;
}

// ---------------------------------------------------------------------------
// fp32 -> bf16 conversion (grid-stride)
// ---------------------------------------------------------------------------
__global__ void f2bf_kernel(const float* __restrict__ in,
                            unsigned short* __restrict__ out, long n) {
    long i = (long)blockIdx.x * blockDim.x + threadIdx.x;
    long stride = (long)gridDim.x * blockDim.x;
    for (; i < n; i += stride) out[i] = f2b(in[i]);
}

#define LDT 40   // padded LDS row stride (halves): 80B, 16B aligned, bank-friendly

// ---------------------------------------------------------------------------
// TDM: DMA a 2D bf16 tile [tile_d1 rows x tile_d0 cols] from global to LDS.
// Hardware pads LDS by 4 DWORDs (8 halves) after every 16 DWORDs (32 halves)
// -> reproduces the LDT=40 padded row stride. Rows/cols beyond
// tensor_d1/tensor_d0 (remaining extent from the tile origin) read as zero.
// Descriptor bit layout per CDNA5 ISA 8.3/8.4 (D# groups 0 and 1).
// This toolchain's builtin takes 6 args: (g0 v4u, g1 v8i, g2 v4i, g3 v4i,
// extra v8i, cpol i32); groups 2/3 are unused for 2D tiles -> zeros.
// ---------------------------------------------------------------------------
#if __has_builtin(__builtin_amdgcn_tensor_load_to_lds)
#define USE_TDM 1
__device__ __forceinline__ void tdm_load_2d(unsigned lds_off,
                                            const unsigned short* gptr,
                                            unsigned tensor_d0, unsigned tensor_d1,
                                            unsigned tile_d0, unsigned tile_d1,
                                            unsigned stride_d0 /* elements */) {
    unsigned long long ga = (unsigned long long)(uintptr_t)gptr;
    v4u g0;
    g0[0] = 1u;                                             // count=1, user D#
    g0[1] = lds_off;                                        // lds_addr (bytes)
    g0[2] = (unsigned)(ga & 0xffffffffu);                   // global_addr[31:0]
    g0[3] = (unsigned)((ga >> 32) & 0x1ffffffu) | (2u << 30); // addr[56:32], type=2
    v8i g1;
    g1[0] = (int)((1u << 16)      // data_size = 1 -> 2 bytes
                | (1u << 20)      // pad_enable
                | (3u << 22)      // pad_interval code 3 -> every 16 DWORDs
                | (3u << 25));    // pad_amount  code 3 -> 4 DWORDs (8 halves)
    g1[1] = (int)((tensor_d0 & 0xffffu) << 16);                              // dim0[15:0]
    g1[2] = (int)(((tensor_d0 >> 16) & 0xffffu) | ((tensor_d1 & 0xffffu) << 16)); // dim0[31:16], dim1[15:0]
    g1[3] = (int)(((tensor_d1 >> 16) & 0xffffu) | ((tile_d0 & 0xffffu) << 16));   // dim1[31:16], tile_dim0
    g1[4] = (int)(tile_d1 & 0xffffu);                                        // tile_dim1 (tile_dim2=0)
    g1[5] = (int)stride_d0;                                                  // dim0_stride[31:0]
    g1[6] = 0;                                                               // stride bits, dim1_stride=0
    g1[7] = 0;
    v4i z4i = {0, 0, 0, 0};
    v8i z8i = {0, 0, 0, 0, 0, 0, 0, 0};
    __builtin_amdgcn_tensor_load_to_lds(g0, g1, z4i, z4i, z8i, 0);
}
#else
#define USE_TDM 0
#endif

// ---------------------------------------------------------------------------
// C[M,N] = A[M,K](bf16) @ W[N,K]^T(bf16)   (row-major, K contiguous on both)
// mode: 0 = none, 1 = *KSC, 2 = *QSC*softplus(pds[col%128])
// outF32: 1 -> float output, 0 -> bf16 output
// block = 256 threads (8 waves); tile BM=128 x BN=128, BK=32, double-buffered.
// ---------------------------------------------------------------------------
__global__ __launch_bounds__(256) void gemm_bf16_kernel(
    const unsigned short* __restrict__ A,
    const unsigned short* __restrict__ W,
    void* __restrict__ Cout,
    int M, int N, int K, int mode,
    const float* __restrict__ pds, int outF32)
{
    __shared__ unsigned short As[2][128 * LDT];
    __shared__ unsigned short Bs[2][128 * LDT];

    const int tid  = threadIdx.x;
    const int lane = tid & 31;
    const int wv   = tid >> 5;           // wave id 0..7 -> M sub-strip
    const int hi   = lane >> 4;          // lane group (0: lanes 0-15, 1: 16-31)
    const int ln16 = lane & 15;
    const int bm   = blockIdx.x * 128;
    const int bn   = blockIdx.y * 128;
    const bool issuer = (tid < 32);      // wave 0 drives the TDM

    v8f acc[8];
#pragma unroll
    for (int t = 0; t < 8; ++t) acc[t] = {};

#if USE_TDM
    // prologue: DMA k-step 0 into buffer 0
    if (issuer) {
        tdm_load_2d((unsigned)(uintptr_t)&As[0][0], A + (size_t)bm * K,
                    (unsigned)K, (unsigned)(M - bm), 32u, 128u, (unsigned)K);
        tdm_load_2d((unsigned)(uintptr_t)&Bs[0][0], W + (size_t)bn * K,
                    (unsigned)K, (unsigned)(N - bn), 32u, 128u, (unsigned)K);
        __builtin_amdgcn_s_wait_tensorcnt(0);
    }
    __syncthreads();
#endif

    int buf = 0;
    for (int kb = 0; kb < K; kb += 32) {
#if USE_TDM
        // prefetch next k-step into the other buffer while we compute
        if (issuer && (kb + 32 < K)) {
            tdm_load_2d((unsigned)(uintptr_t)&As[buf ^ 1][0],
                        A + (size_t)bm * K + (kb + 32),
                        (unsigned)(K - (kb + 32)), (unsigned)(M - bm), 32u, 128u, (unsigned)K);
            tdm_load_2d((unsigned)(uintptr_t)&Bs[buf ^ 1][0],
                        W + (size_t)bn * K + (kb + 32),
                        (unsigned)(K - (kb + 32)), (unsigned)(N - bn), 32u, 128u, (unsigned)K);
        }
#else
        __syncthreads();
        for (int c = tid; c < 512; c += 256) {
            int r = c >> 2, off = (c & 3) * 8;
            int row = bm + r;
            uint4 av = make_uint4(0, 0, 0, 0);
            if (row < M) av = *(const uint4*)(A + (size_t)row * K + kb + off);
            *(uint4*)(&As[buf][0] + r * LDT + off) = av;
            uint4 bv = *(const uint4*)(W + (size_t)(bn + r) * K + kb + off);
            *(uint4*)(&Bs[buf][0] + r * LDT + off) = bv;
        }
        __syncthreads();
#endif
        const unsigned short* Asb = &As[buf][0];
        const unsigned short* Bsb = &Bs[buf][0];

        // A fragment (16x32): lane ln16 = row M; lanes 0-15 K={0..15}, 16-31 K={16..31}
        const int ar = wv * 16 + ln16;
        uint4 a0 = *(const uint4*)(Asb + ar * LDT + hi * 16);
        uint4 a1 = *(const uint4*)(Asb + ar * LDT + hi * 16 + 8);
        v16bf af = mkfrag(a0, a1);

        // preload all B fragments, then issue the WMMA chain back-to-back
        uint4 bb[16];
#pragma unroll
        for (int t = 0; t < 8; ++t) {
            const int br = t * 16 + ln16;
            bb[2 * t]     = *(const uint4*)(Bsb + br * LDT + hi * 16);
            bb[2 * t + 1] = *(const uint4*)(Bsb + br * LDT + hi * 16 + 8);
        }
#pragma unroll
        for (int t = 0; t < 8; ++t)
            acc[t] = __builtin_amdgcn_wmma_f32_16x16x32_bf16(
                false, af, false, mkfrag(bb[2 * t], bb[2 * t + 1]),
                (short)0, acc[t], false, false);

#if USE_TDM
        if (issuer) __builtin_amdgcn_s_wait_tensorcnt(0);
        __syncthreads();
#endif
        buf ^= 1;
    }

    // epilogue
#pragma unroll
    for (int t = 0; t < 8; ++t) {
        const int col = bn + t * 16 + ln16;
        float s = 1.0f;
        if (mode == 1)      s = KSC;
        else if (mode == 2) s = QSC * log1pf(__expf(pds[col & (DHEAD - 1)]));
#pragma unroll
        for (int vv = 0; vv < 8; ++vv) {
            const int row = bm + wv * 16 + hi * 8 + vv;
            if (row < M) {
                float x = acc[t][vv] * s;
                if (outF32)
                    ((float*)Cout)[(size_t)row * N + col] = x;
                else
                    ((unsigned short*)Cout)[(size_t)row * N + col] = f2b(x);
            }
        }
    }
}

// ---------------------------------------------------------------------------
// Chunked relative-position attention.
// One wave per (batch, block, head). 128 threads = 4 waves per WG;
// grid = BATCH * NB * (NHEAD/4).
// scores: Q[12,128] Kctx[24,128]^T and Q relk[25,128]^T via WMMA (16 ops),
// rel-shift + softcap + softmax scalar per row, out = P @ Vctx via WMMA (8 ops).
// ---------------------------------------------------------------------------
__global__ __launch_bounds__(128) void attn_kernel(
    const unsigned short* __restrict__ q,
    const unsigned short* __restrict__ k,
    const unsigned short* __restrict__ vbuf,
    const unsigned short* __restrict__ relk,
    unsigned short* __restrict__ o)
{
    __shared__ unsigned short VT[4][DHEAD * LDT];   // V^T, [d][j] padded
    __shared__ float          acT[4][16 * 32];
    __shared__ float          bdT[4][16 * 32];
    __shared__ unsigned short pT[4][16 * LDT];      // probs, A-operand layout

    const int lane = threadIdx.x & 31;
    const int w    = threadIdx.x >> 5;
    const int hi   = lane >> 4;
    const int ln16 = lane & 15;

    const int gid = blockIdx.x;
    const int h   = (gid & 1) * 4 + w;
    const int blk = (gid >> 1) % NB;
    const int b   = gid / (2 * NB);

    const uint4 z4 = make_uint4(0, 0, 0, 0);

    // ---- stage V^T into LDS (transpose scatter) ----
    for (int t = lane; t < CTX * 16; t += 32) {
        int j = t >> 4, c8 = (t & 15) * 8;
        int seq = blk * CHUNK - (CTX - CHUNK) + j;
        uint4 x = z4;
        if (seq >= 0 && seq < S_LEN)
            x = *(const uint4*)(vbuf + (((size_t)(b * S_LEN + seq)) * NHEAD + h) * DHEAD + c8);
        const unsigned short* xs = (const unsigned short*)&x;
#pragma unroll
        for (int i = 0; i < 8; ++i) VT[w][(c8 + i) * LDT + j] = xs[i];
    }
    for (int t = lane; t < DHEAD * 8; t += 32) {
        int d = t >> 3, j = CTX + (t & 7);
        VT[w][d * LDT + j] = 0;
    }

    // ---- content + positional scores via WMMA ----
    v8f ac0 = {}, ac1 = {}, bd0 = {}, bd1 = {};
    const int mq = ln16;
    const bool qvalid = mq < CHUNK;
    size_t qbase = 0;
    if (qvalid) qbase = (((size_t)(b * S_LEN + blk * CHUNK + mq)) * NHEAD + h) * DHEAD;

#pragma unroll
    for (int ds = 0; ds < 4; ++ds) {           // K-dim (head dim) in steps of 32
        uint4 a0 = z4, a1 = z4;
        if (qvalid) {
            a0 = *(const uint4*)(q + qbase + ds * 32 + hi * 16);
            a1 = *(const uint4*)(q + qbase + ds * 32 + hi * 16 + 8);
        }
        v16bf af = mkfrag(a0, a1);

#pragma unroll
        for (int tile = 0; tile < 2; ++tile) {  // key context tiles (j 0-15, 16-31)
            int j = tile * 16 + ln16;
            int seq = blk * CHUNK - (CTX - CHUNK) + j;
            uint4 b0 = z4, b1 = z4;
            if (j < CTX && seq >= 0 && seq < S_LEN) {
                const unsigned short* kp =
                    k + (((size_t)(b * S_LEN + seq)) * NHEAD + h) * DHEAD + ds * 32 + hi * 16;
                b0 = *(const uint4*)(kp);
                b1 = *(const uint4*)(kp + 8);
            }
            v8f& acc = tile ? ac1 : ac0;
            acc = __builtin_amdgcn_wmma_f32_16x16x32_bf16(
                false, af, false, mkfrag(b0, b1), (short)0, acc, false, false);
        }
#pragma unroll
        for (int tile = 0; tile < 2; ++tile) {  // relative-position tiles (p 0-15, 16-24)
            int p = tile * 16 + ln16;
            uint4 b0 = z4, b1 = z4;
            if (p < CTX + 1) {
                const unsigned short* rp =
                    relk + ((size_t)p * NHEAD + h) * DHEAD + ds * 32 + hi * 16;
                b0 = *(const uint4*)(rp);
                b1 = *(const uint4*)(rp + 8);
            }
            v8f& acc = tile ? bd1 : bd0;
            acc = __builtin_amdgcn_wmma_f32_16x16x32_bf16(
                false, af, false, mkfrag(b0, b1), (short)0, acc, false, false);
        }
    }

    // ---- spill score fragments to LDS (C layout: lane=col, vgpr=row) ----
#pragma unroll
    for (int vv = 0; vv < 8; ++vv) {
        int m = hi * 8 + vv;
        acT[w][m * 32 + ln16]      = ac0[vv];
        acT[w][m * 32 + 16 + ln16] = ac1[vv];
        bdT[w][m * 32 + ln16]      = bd0[vv];
        bdT[w][m * 32 + 16 + ln16] = bd1[vv];
    }
    __syncthreads();

    // ---- rel-shift + softcap + softmax, one row per lane (lanes 0-15) ----
    if (lane < 16) {
        const int c = lane;
        float vals[CTX];
        float mx = -3.4e38f;
#pragma unroll
        for (int j = 0; j < CTX; ++j) {
            float bdv = (j >= c) ? bdT[w][c * 32 + (j - c)]
                                 : bdT[w][(c - 1) * 32 + (CTX + 1 - c + j)];
            float l = acT[w][c * 32 + j] + bdv;
            l = tanhf(l * 0.02f) * 50.0f;          // softcap 50
            vals[j] = l;
            mx = fmaxf(mx, l);
        }
        float sum = 0.0f;
#pragma unroll
        for (int j = 0; j < CTX; ++j) { float e = __expf(vals[j] - mx); vals[j] = e; sum += e; }
        float inv = 1.0f / sum;
#pragma unroll
        for (int j = 0; j < CTX; ++j) pT[w][c * LDT + j] = f2b(vals[j] * inv);
#pragma unroll
        for (int j = CTX; j < 32; ++j) pT[w][c * LDT + j] = 0;
    }
    __syncthreads();

    // ---- out = P[16,32] @ V[32,128] via WMMA over 8 N-tiles ----
    uint4 p0 = *(const uint4*)(pT[w] + ln16 * LDT + hi * 16);
    uint4 p1 = *(const uint4*)(pT[w] + ln16 * LDT + hi * 16 + 8);
    v16bf pf = mkfrag(p0, p1);

#pragma unroll
    for (int t = 0; t < 8; ++t) {
        int d = t * 16 + ln16;
        uint4 b0 = *(const uint4*)(VT[w] + d * LDT + hi * 16);
        uint4 b1 = *(const uint4*)(VT[w] + d * LDT + hi * 16 + 8);
        v8f oacc = {};
        oacc = __builtin_amdgcn_wmma_f32_16x16x32_bf16(
            false, pf, false, mkfrag(b0, b1), (short)0, oacc, false, false);
#pragma unroll
        for (int vv = 0; vv < 8; ++vv) {
            int m = hi * 8 + vv;
            if (m < CHUNK)
                o[(((size_t)(b * S_LEN + blk * CHUNK + m)) * NHEAD + h) * DHEAD + d] =
                    f2b(oacc[vv]);
        }
    }
}

// ---------------------------------------------------------------------------
// Host launcher. Workspace layout (bf16 buffers), ~263 MB total:
//   hidden, q, k, v, attn_out : 24576*1024*2 bytes each
//   Wq,Wk,Wv,Wrel,Wpost bf16  : 1024*1024*2 each; pos_emb, rel_k : 25*1024*2
// ---------------------------------------------------------------------------
extern "C" void kernel_launch(void* const* d_in, const int* in_sizes, int n_in,
                              void* d_out, int out_size, void* d_ws, size_t ws_size,
                              hipStream_t stream) {
    const float* hidden = (const float*)d_in[0];
    const float* pos    = (const float*)d_in[1];
    const float* Wq     = (const float*)d_in[2];
    const float* Wk     = (const float*)d_in[3];
    const float* Wv     = (const float*)d_in[4];
    const float* Wpost  = (const float*)d_in[5];
    const float* Wrel   = (const float*)d_in[6];
    const float* pds    = (const float*)d_in[7];

    char* ws = (char*)d_ws;
    size_t off = 0;
    auto take = [&](size_t bytes) -> unsigned short* {
        size_t o = off;
        off += (bytes + 255) & ~(size_t)255;
        return (unsigned short*)(ws + o);
    };

    const size_t actB = (size_t)MTOT * HIDDIM * 2;
    const size_t wB   = (size_t)HIDDIM * HIDDIM * 2;

    unsigned short* hbf   = take(actB);
    unsigned short* qbf   = take(actB);
    unsigned short* kbf   = take(actB);
    unsigned short* vbf   = take(actB);
    unsigned short* obf   = take(actB);
    unsigned short* wqb   = take(wB);
    unsigned short* wkb   = take(wB);
    unsigned short* wvb   = take(wB);
    unsigned short* wrelb = take(wB);
    unsigned short* wpob  = take(wB);
    unsigned short* peb   = take((size_t)25 * HIDDIM * 2);
    unsigned short* relkb = take((size_t)25 * HIDDIM * 2);

    // 1) fp32 -> bf16
    f2bf_kernel<<<8192, 256, 0, stream>>>(hidden, hbf, (long)MTOT * HIDDIM);
    f2bf_kernel<<<1024, 256, 0, stream>>>(Wq, wqb, (long)HIDDIM * HIDDIM);
    f2bf_kernel<<<1024, 256, 0, stream>>>(Wk, wkb, (long)HIDDIM * HIDDIM);
    f2bf_kernel<<<1024, 256, 0, stream>>>(Wv, wvb, (long)HIDDIM * HIDDIM);
    f2bf_kernel<<<1024, 256, 0, stream>>>(Wrel, wrelb, (long)HIDDIM * HIDDIM);
    f2bf_kernel<<<1024, 256, 0, stream>>>(Wpost, wpob, (long)HIDDIM * HIDDIM);
    f2bf_kernel<<<32, 256, 0, stream>>>(pos, peb, (long)25 * HIDDIM);

    // 2) projections (WMMA GEMMs, TDM-staged)
    dim3 g(MTOT / 128, HIDDIM / 128);
    gemm_bf16_kernel<<<g, 256, 0, stream>>>(hbf, wqb, (void*)qbf, MTOT, HIDDIM, HIDDIM, 2, pds, 0);
    gemm_bf16_kernel<<<g, 256, 0, stream>>>(hbf, wkb, (void*)kbf, MTOT, HIDDIM, HIDDIM, 1, nullptr, 0);
    gemm_bf16_kernel<<<g, 256, 0, stream>>>(hbf, wvb, (void*)vbf, MTOT, HIDDIM, HIDDIM, 0, nullptr, 0);
    gemm_bf16_kernel<<<dim3(1, HIDDIM / 128), 256, 0, stream>>>(
        peb, wrelb, (void*)relkb, 25, HIDDIM, HIDDIM, 0, nullptr, 0);

    // 3) chunked attention
    attn_kernel<<<BATCH * NB * (NHEAD / 4), 128, 0, stream>>>(qbf, kbf, vbf, relkb, obf);

    // 4) output projection -> fp32 d_out
    gemm_bf16_kernel<<<g, 256, 0, stream>>>(obf, wpob, d_out, MTOT, HIDDIM, HIDDIM, 0, nullptr, 1);
}
